// whole_network_38371237822511
// MI455X (gfx1250) — matrix operand order repaired and checked by reading, not compiled, not run on hardware
//
#include <hip/hip_runtime.h>
#include <math.h>

typedef __attribute__((ext_vector_type(16))) _Float16 v16h;
typedef __attribute__((ext_vector_type(8)))  _Float16 v8h;
typedef __attribute__((ext_vector_type(8)))  float    v8f;

#define NB 512      // batch
#define NR 130      // ROIs
#define NH 8        // heads
#define QD 64       // per-head dim
#define NBOLD 200
#define BRROWS (NB * NR)   // 66560

// gfx1250 async global->LDS copies (ASYNCcnt path), guarded so the build
// cannot regress if the toolchain lacks the builtins.
#if defined(__has_builtin)
#  if __has_builtin(__builtin_amdgcn_global_load_async_to_lds_b128) && \
      __has_builtin(__builtin_amdgcn_s_wait_asynccnt)
#    define USE_ASYNC_LDS 1
#  endif
#endif
#ifndef USE_ASYNC_LDS
#  define USE_ASYNC_LDS 0
#endif

#if USE_ASYNC_LDS
// builtin signature (from compiler diagnostic): params are int4* in
// addrspace(1) (global src) and addrspace(3) (LDS dst), then imm offset, cpol.
typedef int ai4 __attribute__((ext_vector_type(4)));
typedef __attribute__((address_space(1))) ai4 ga_i4;   // global int4
typedef __attribute__((address_space(3))) ai4 ls_i4;   // LDS int4
#define ASYNC_CP16(gptr, lptr)                                               \
  __builtin_amdgcn_global_load_async_to_lds_b128((ga_i4*)(gptr),             \
                                                 (ls_i4*)(lptr), 0, 0)
#endif

// ---------------------------------------------------------------------------
// 0) NaN clean (vectorized; element count is a multiple of 4)
// ---------------------------------------------------------------------------
__global__ void nan_clean_kernel(const float4* __restrict__ x,
                                 float4* __restrict__ y, int n4) {
  int i = blockIdx.x * 256 + threadIdx.x;
  if (i < n4) {
    float4 v = x[i];
    v.x = (v.x == v.x) ? v.x : 0.0f;
    v.y = (v.y == v.y) ? v.y : 0.0f;
    v.z = (v.z == v.z) ? v.z : 0.0f;
    v.w = (v.w == v.w) ? v.w : 0.0f;
    y[i] = v;
  }
}

// ---------------------------------------------------------------------------
// 1) Generic batched WMMA GEMM: C = A[M,K] x B (+bias), f16 in / f32 acc.
//    b_trans==1: B is W[N,K] row-major (A @ W^T); b_trans==0: B is [K,N].
//    64x64 block tile, K-step 32, 8 waves x two 16x16 C tiles,
//    register-staged + software-pipelined global->LDS path.
// ---------------------------------------------------------------------------
__global__ void gemm_wmma_kernel(const float* __restrict__ A,
                                 const float* __restrict__ Bm,
                                 const float* __restrict__ bias,
                                 float* __restrict__ Cf,
                                 _Float16* __restrict__ Ch,
                                 int M, int N, int K,
                                 long long sA, long long sB, long long sC,
                                 int b_trans, int relu_flag, int out_half) {
  __shared__ __align__(16) _Float16 As[64][40];   // [m][k], row = 80B
  __shared__ __align__(16) _Float16 Bs[64][40];   // [n][k] N-major

  const int tid  = threadIdx.x;
  const int lane = tid & 31;
  const int wave = tid >> 5;
  const int hid  = lane >> 4;
  const int lm   = lane & 15;
  const int wr   = wave >> 1;      // 0..3 -> m tile
  const int wc   = wave & 1;       // 0..1 -> 32-wide n slab

  const int m0 = blockIdx.x * 64;
  const int n0 = blockIdx.y * 64;
  const size_t oA = (size_t)blockIdx.z * (size_t)sA;
  const size_t oB = (size_t)blockIdx.z * (size_t)sB;
  const size_t oC = (size_t)blockIdx.z * (size_t)sC;

  const bool k4 = (K & 3) == 0;
  const bool n4 = (N & 3) == 0;

  const int ar  = tid >> 2;            // row / n (0..63)
  const int ac8 = (tid & 3) << 3;      // k segment start
  const int bk  = tid >> 3;            // k (0..31)  [b_trans==0 mapping]
  const int bn8 = (tid & 7) << 3;      // n segment start [b_trans==0]

  float fa[8], fb[8];

  auto loadA = [&](int k0) {
    const int gm = m0 + ar;
    const int gk = k0 + ac8;
    if (gm < M && gk + 7 < K) {
      const float* ap = A + oA + (size_t)gm * K + gk;
      if (k4) {
        float4 u = *(const float4*)ap;
        float4 w = *(const float4*)(ap + 4);
        fa[0] = u.x; fa[1] = u.y; fa[2] = u.z; fa[3] = u.w;
        fa[4] = w.x; fa[5] = w.y; fa[6] = w.z; fa[7] = w.w;
      } else {
#pragma unroll
        for (int j = 0; j < 8; ++j) fa[j] = ap[j];
      }
    } else {
#pragma unroll
      for (int j = 0; j < 8; ++j)
        fa[j] = (gm < M && gk + j < K) ? A[oA + (size_t)gm * K + gk + j] : 0.0f;
    }
  };

  auto loadB = [&](int k0) {
    if (b_trans) {
      const int gn = n0 + ar;
      const int gk = k0 + ac8;
      if (gn < N && gk + 7 < K) {
        const float* bp = Bm + oB + (size_t)gn * K + gk;
        if (k4) {
          float4 u = *(const float4*)bp;
          float4 w = *(const float4*)(bp + 4);
          fb[0] = u.x; fb[1] = u.y; fb[2] = u.z; fb[3] = u.w;
          fb[4] = w.x; fb[5] = w.y; fb[6] = w.z; fb[7] = w.w;
        } else {
#pragma unroll
          for (int j = 0; j < 8; ++j) fb[j] = bp[j];
        }
      } else {
#pragma unroll
        for (int j = 0; j < 8; ++j)
          fb[j] = (gn < N && gk + j < K) ? Bm[oB + (size_t)gn * K + gk + j] : 0.0f;
      }
    } else {
      const int gk = k0 + bk;
      const int gn = n0 + bn8;
      if (gk < K && gn + 7 < N) {
        const float* bp = Bm + oB + (size_t)gk * N + gn;   // contiguous in n
        if (n4) {
          float4 u = *(const float4*)bp;
          float4 w = *(const float4*)(bp + 4);
          fb[0] = u.x; fb[1] = u.y; fb[2] = u.z; fb[3] = u.w;
          fb[4] = w.x; fb[5] = w.y; fb[6] = w.z; fb[7] = w.w;
        } else {
#pragma unroll
          for (int j = 0; j < 8; ++j) fb[j] = bp[j];
        }
      } else {
#pragma unroll
        for (int j = 0; j < 8; ++j)
          fb[j] = (gk < K && gn + j < N) ? Bm[oB + (size_t)gk * N + gn + j] : 0.0f;
      }
    }
  };

  auto stage = [&]() {
    v8h ha;
#pragma unroll
    for (int j = 0; j < 8; ++j) ha[j] = (_Float16)fa[j];
    *(v8h*)&As[ar][ac8] = ha;                       // ds_store_b128
    if (b_trans) {
      v8h hb;
#pragma unroll
      for (int j = 0; j < 8; ++j) hb[j] = (_Float16)fb[j];
      *(v8h*)&Bs[ar][ac8] = hb;
    } else {
#pragma unroll
      for (int j = 0; j < 8; ++j) Bs[bn8 + j][bk] = (_Float16)fb[j];
    }
  };

  v8f acc0 = {};
  v8f acc1 = {};

  loadA(0); loadB(0);
  for (int k0 = 0; k0 < K; k0 += 32) {
    stage();
    __syncthreads();                  // LDS tile ready
    if (k0 + 32 < K) { loadA(k0 + 32); loadB(k0 + 32); }   // overlap with math

    v16h a, b0v, b1v;
#pragma unroll
    for (int e = 0; e < 16; ++e) {
      int ka = e + (hid << 3) + ((e & 8) ? 8 : 0);   // A 16x32 layout
      a[e]   = As[wr * 16 + lm][ka];
      int kb = e + (hid << 3) + ((e & 8) ? 8 : 0);   // B 32x16 layout (same k order as A)
      b0v[e] = Bs[wc * 32 + lm][kb];
      b1v[e] = Bs[wc * 32 + 16 + lm][kb];
    }
    acc0 = __builtin_amdgcn_wmma_f32_16x16x32_f16(false, a, false, b0v,
                                                  (short)0, acc0, false, false);
    acc1 = __builtin_amdgcn_wmma_f32_16x16x32_f16(false, a, false, b1v,
                                                  (short)0, acc1, false, false);
    __syncthreads();                  // done reading before next stage()
  }

  // epilogue: C layout row = v + 8*hid, col = lm
#pragma unroll
  for (int v = 0; v < 8; ++v) {
    int m = m0 + wr * 16 + v + (hid << 3);
    if (m >= M) continue;
#pragma unroll
    for (int j = 0; j < 2; ++j) {
      int n = n0 + wc * 32 + j * 16 + lm;
      if (n >= N) continue;
      float val = (j == 0) ? acc0[v] : acc1[v];
      if (bias) val += bias[n];
      if (relu_flag && val < 0.0f) val = 0.0f;
      size_t off = oC + (size_t)m * N + n;
      if (out_half) Ch[off] = (_Float16)val;
      else          Cf[off] = val;
    }
  }
}

// ---------------------------------------------------------------------------
// 2) Fused attention: WMMA scores -> softmax -> relu(p-0.01) -> head fusion.
//    One block per (16-row tile, batch); rel never hits HBM.
//    K/Q tiles are pure f16 copies -> async global->LDS (ASYNCcnt) when
//    available; pad rows zero-filled once (async can't zero-fill).
// ---------------------------------------------------------------------------
__global__ void attention_kernel(const _Float16* __restrict__ Qh,
                                 const _Float16* __restrict__ Kb,
                                 const float* __restrict__ fus_w,
                                 const float* __restrict__ fus_b,
                                 float* __restrict__ Aout) {
  __shared__ __align__(16) _Float16 Ks[144][80];  // 160B row stride
  __shared__ __align__(16) _Float16 Qs[16][80];
  __shared__ float Ss[16][144];
  __shared__ float Ar[16][132];
  __shared__ float red[16][16];

  const int r0   = blockIdx.x * 16;
  const int b    = blockIdx.y;
  const int tid  = threadIdx.x;
  const int lane = tid & 31;
  const int wave = tid >> 5;
  const int hid  = lane >> 4;
  const int lm   = lane & 15;
  const int row  = tid >> 4;   // softmax: 16 threads per row
  const int sub  = tid & 15;

  const int qrows = (NR - r0 < 16) ? (NR - r0) : 16;

  // one-time zero fill of K/Q tiles (pad rows persist across heads)
  for (int i = tid; i < 1152 + 128; i += 256) {
    v8h z;
#pragma unroll
    for (int j = 0; j < 8; ++j) z[j] = (_Float16)0.0f;
    if (i < 1152) *(v8h*)&Ks[i >> 3][(i & 7) << 3] = z;
    else {
      int q = i - 1152;
      *(v8h*)&Qs[q >> 3][(q & 7) << 3] = z;
    }
  }
  for (int i = tid; i < 16 * 132; i += 256) Ar[i / 132][i % 132] = 0.0f;
  __syncthreads();

  for (int h = 0; h < NH; ++h) {
    // stage K[b, s<130, h, :] as 16B chunks (130 rows x 8 chunks)
    for (int i = tid; i < NR * 8; i += 256) {
      int s = i >> 3, q8 = (i & 7) << 3;
      const _Float16* g = &Kb[((size_t)(b * NR + s) * NH + h) * QD + q8];
#if USE_ASYNC_LDS
      ASYNC_CP16(g, &Ks[s][q8]);
#else
      *(v8h*)&Ks[s][q8] = *(const v8h*)g;
#endif
    }
    // stage valid Q rows (qrows x 8 chunks)
    for (int i = tid; i < qrows * 8; i += 256) {
      int rr = i >> 3, q8 = (i & 7) << 3;
      const _Float16* g = &Qh[((size_t)(b * NR + r0 + rr) * NH + h) * QD + q8];
#if USE_ASYNC_LDS
      ASYNC_CP16(g, &Qs[rr][q8]);
#else
      *(v8h*)&Qs[rr][q8] = *(const v8h*)g;
#endif
    }
#if USE_ASYNC_LDS
    __builtin_amdgcn_s_wait_asynccnt(0);
#endif
    __syncthreads();

    // A-fragments for both 32-wide k-steps (shared across column tiles)
    v16h afr0, afr1;
#pragma unroll
    for (int e = 0; e < 16; ++e) {
      int kl = e + (hid << 4);
      afr0[e] = Qs[lm][kl];
      afr1[e] = Qs[lm][32 + kl];
    }
    // 9 column tiles over 8 waves
    for (int ct = wave; ct < 9; ct += 8) {
      v8f acc = {};
      v16h bfr;
#pragma unroll
      for (int e = 0; e < 16; ++e) bfr[e] = Ks[ct * 16 + lm][e + (hid << 4)];
      acc = __builtin_amdgcn_wmma_f32_16x16x32_f16(false, afr0, false, bfr,
                                                   (short)0, acc, false, false);
#pragma unroll
      for (int e = 0; e < 16; ++e) bfr[e] = Ks[ct * 16 + lm][32 + e + (hid << 4)];
      acc = __builtin_amdgcn_wmma_f32_16x16x32_f16(false, afr1, false, bfr,
                                                   (short)0, acc, false, false);
#pragma unroll
      for (int vv = 0; vv < 8; ++vv)
        Ss[vv + (hid << 3)][ct * 16 + lm] = acc[vv] * (1.0f / 64.0f);
    }
    __syncthreads();

    // parallel softmax: 16 threads per row, LDS tree reductions
    float mx = -3.4e38f;
    for (int s = sub; s < NR; s += 16) mx = fmaxf(mx, Ss[row][s]);
    red[row][sub] = mx;
    __syncthreads();
#pragma unroll
    for (int st = 8; st > 0; st >>= 1) {
      if (sub < st) red[row][sub] = fmaxf(red[row][sub], red[row][sub + st]);
      __syncthreads();
    }
    float rmax = red[row][0];
    __syncthreads();
    float sm = 0.0f;
    for (int s = sub; s < NR; s += 16) sm += __expf(Ss[row][s] - rmax);
    red[row][sub] = sm;
    __syncthreads();
#pragma unroll
    for (int st = 8; st > 0; st >>= 1) {
      if (sub < st) red[row][sub] += red[row][sub + st];
      __syncthreads();
    }
    float inv = 1.0f / red[row][0];
    float fw  = fus_w[h];
    for (int s = sub; s < NR; s += 16) {
      float pv = __expf(Ss[row][s] - rmax) * inv - 0.01f;
      Ar[row][s] += fw * fmaxf(pv, 0.0f);
    }
    __syncthreads();   // protect Ks/Qs/Ss/red before next head
  }

  float fb = fus_b[0];
  for (int i = tid; i < 16 * NR; i += 256) {
    int rr = i / NR, s = i % NR;
    int gr = r0 + rr;
    if (gr < NR)
      Aout[(size_t)b * (NR * NR) + (size_t)gr * NR + s] = Ar[rr][s] + fb;
  }
}

// ---------------------------------------------------------------------------
// 3) BatchNorm over (B,F) per ROI channel + ReLU + optional residual; in-place
// ---------------------------------------------------------------------------
__global__ void bn_relu_res_kernel(const float* __restrict__ h,
                                   const float* __restrict__ res,
                                   float* __restrict__ out,
                                   const float* __restrict__ gamma,
                                   const float* __restrict__ beta,
                                   int F, int useRes) {
  const int r = blockIdx.x;               // 0..129
  const int tid = threadIdx.x;
  const int Ntot = NB * F;
  __shared__ float sh_s[256], sh_q[256];

  float s = 0.0f, q = 0.0f;
  for (int i = tid; i < Ntot; i += 256) {
    int bb = i / F, f = i - bb * F;
    float v = h[((size_t)bb * NR + r) * F + f];
    s += v; q += v * v;
  }
  sh_s[tid] = s; sh_q[tid] = q;
  __syncthreads();
  for (int st = 128; st > 0; st >>= 1) {
    if (tid < st) { sh_s[tid] += sh_s[tid + st]; sh_q[tid] += sh_q[tid + st]; }
    __syncthreads();
  }
  float mean = sh_s[0] / (float)Ntot;
  float var  = sh_q[0] / (float)Ntot - mean * mean;
  float scale = rsqrtf(var + 1e-5f) * gamma[r];
  float shift = beta[r] - mean * scale;

  for (int i = tid; i < Ntot; i += 256) {
    int bb = i / F, f = i - bb * F;
    size_t idx = ((size_t)bb * NR + r) * F + f;
    float v = h[idx] * scale + shift;
    v = fmaxf(v, 0.0f);
    if (useRes) v += res[idx];
    out[idx] = v;
  }
}

// ---------------------------------------------------------------------------
// 4) final 2-class softmax
// ---------------------------------------------------------------------------
__global__ void softmax2_kernel(const float* __restrict__ z,
                                float* __restrict__ out) {
  int i = blockIdx.x * 256 + threadIdx.x;
  if (i < NB) {
    float a = z[i * 2 + 0], b = z[i * 2 + 1];
    float m = fmaxf(a, b);
    float ea = __expf(a - m), eb = __expf(b - m);
    float inv = 1.0f / (ea + eb);
    out[i * 2 + 0] = ea * inv;
    out[i * 2 + 1] = eb * inv;
  }
}

// ---------------------------------------------------------------------------
static inline void gemm(hipStream_t st, const float* A, const float* B,
                        const float* bias, float* Cf, _Float16* Ch,
                        int M, int N, int K,
                        long long sA, long long sB, long long sC,
                        int b_trans, int relu, int out_half, int Z) {
  dim3 grid((M + 63) / 64, (N + 63) / 64, Z);
  gemm_wmma_kernel<<<grid, dim3(256), 0, st>>>(A, B, bias, Cf, Ch, M, N, K,
                                               sA, sB, sC, b_trans, relu, out_half);
}

extern "C" void kernel_launch(void* const* d_in, const int* in_sizes, int n_in,
                              void* d_out, int out_size, void* d_ws, size_t ws_size,
                              hipStream_t stream) {
  (void)in_sizes; (void)n_in; (void)out_size; (void)ws_size;

  const float* x     = (const float*)d_in[0];
  const float* lq_w  = (const float*)d_in[1];
  const float* lq_b  = (const float*)d_in[2];
  const float* lk_w  = (const float*)d_in[3];
  const float* lk_b  = (const float*)d_in[4];
  const float* fus_w = (const float*)d_in[5];
  const float* fus_b = (const float*)d_in[6];
  const float *gW[10], *gB[10], *gG[10], *gBe[10];
  for (int i = 0; i < 10; ++i) {
    gW[i]  = (const float*)d_in[7 + i];
    gB[i]  = (const float*)d_in[17 + i];
    gG[i]  = (const float*)d_in[27 + i];
    gBe[i] = (const float*)d_in[37 + i];
  }
  const float* w1 = (const float*)d_in[47]; const float* b1 = (const float*)d_in[48];
  const float* w2 = (const float*)d_in[49]; const float* b2 = (const float*)d_in[50];
  const float* w3 = (const float*)d_in[51]; const float* b3 = (const float*)d_in[52];

  // workspace layout (float slots; every block 16B aligned)
  float* p = (float*)d_ws;
  float*    x_clean = p;              p += (size_t)BRROWS * NBOLD;
  _Float16* Qh      = (_Float16*)p;   p += (size_t)BRROWS * 512 / 2;
  _Float16* Kh      = (_Float16*)p;   p += (size_t)BRROWS * 512 / 2;
  float*    Aadj    = p;              p += (size_t)NB * NR * NR;
  float*    tmp     = p;              p += (size_t)BRROWS * 256;
  float*    fA      = p;              p += (size_t)BRROWS * 256;
  float*    fB      = p;              p += (size_t)BRROWS * 256;
  float*    z1      = p;              p += (size_t)NB * 512;
  float*    z2      = p;              p += (size_t)NB * 32;
  float*    z3      = p;              p += (size_t)NB * 2;

  // 0) NaN clean
  {
    int n4 = (BRROWS * NBOLD) / 4;
    nan_clean_kernel<<<(n4 + 255) / 256, 256, 0, stream>>>(
        (const float4*)x, (float4*)x_clean, n4);
  }

  // 1) Q/K projections (f16 output): [66560,200] @ [200,512]^T + bias
  gemm(stream, x_clean, lq_w, lq_b, nullptr, Qh, BRROWS, 512, NBOLD,
       0, 0, 0, /*b_trans*/1, /*relu*/0, /*half*/1, 1);
  gemm(stream, x_clean, lk_w, lk_b, nullptr, Kh, BRROWS, 512, NBOLD,
       0, 0, 0, 1, 0, 1, 1);

  // 2) fused attention -> adjacency A [B,130,130]
  attention_kernel<<<dim3(9, NB), 256, 0, stream>>>(Qh, Kh, fus_w, fus_b, Aadj);

  // 3) 10 GCN layers
  static const int GIN[10]  = {200,256,256,128,128,64,64,32,32,16};
  static const int GOUT[10] = {256,256,128,128,64,64,32,32,16,16};
  static const int GRES[10] = {0,1,0,1,0,1,0,1,0,1};
  const float* cur = x_clean;
  float* bufs[2] = {fA, fB};
  for (int i = 0; i < 10; ++i) {
    float* hb = bufs[i & 1];
    // tmp = Adj @ feats  (batched over 512)
    gemm(stream, Aadj, cur, nullptr, tmp, nullptr, NR, GIN[i], NR,
         (long long)NR * NR, (long long)NR * GIN[i], (long long)NR * GIN[i],
         /*b_trans*/0, 0, 0, NB);
    // hb = tmp @ W^T + b
    gemm(stream, tmp, gW[i], gB[i], hb, nullptr, BRROWS, GOUT[i], GIN[i],
         0, 0, 0, 1, 0, 0, 1);
    // BN (per-ROI channel) + ReLU + residual, in place
    bn_relu_res_kernel<<<NR, 256, 0, stream>>>(hb, GRES[i] ? cur : nullptr, hb,
                                               gG[i], gBe[i], GOUT[i], GRES[i]);
    cur = hb;
  }

  // 4) MLP head: [512,2080]->512->32->2 (ReLU each), then softmax
  gemm(stream, cur, w1, b1, z1, nullptr, NB, 512, NR * 16, 0, 0, 0, 1, 1, 0, 1);
  gemm(stream, z1,  w2, b2, z2, nullptr, NB, 32, 512,     0, 0, 0, 1, 1, 0, 1);
  gemm(stream, z2,  w3, b3, z3, nullptr, NB, 2,  32,      0, 0, 0, 1, 1, 0, 1);
  softmax2_kernel<<<2, 256, 0, stream>>>(z3, (float*)d_out);
}
